// CMLInt8Sim_21380347199654
// MI455X (gfx1250) — matrix operand order repaired and checked by reading, not compiled, not run on hardware
//
#include <hip/hip_runtime.h>
#include <math.h>

#define B_DIM   256
#define N_DIM   2048
#define N_MASK  (N_DIM - 1)
#define K_TAPS  5
#define N_STEPS 64

typedef __attribute__((ext_vector_type(16))) _Float16 v16h;
typedef __attribute__((ext_vector_type(8)))  _Float16 v8h;
typedef __attribute__((ext_vector_type(8)))  float    v8f;

// ---- workspace layout (bytes) ----
// Wp   : packed f16 (W_cc + circulant(K_local)) B-fragments = 8388608
// gA/gB: ping-pong grid state 2 * 2097152
static constexpr size_t OFF_WP = 0;
static constexpr size_t OFF_GA = OFF_WP + (size_t)N_DIM * N_DIM * 2;
static constexpr size_t OFF_GB = OFF_GA + (size_t)B_DIM * N_DIM * 4;

// LDS A-tile: 32 rows x 2048 cols f16, rows padded by 8 halves so the
// row stride is 4112B -> lane m of an A-fragment hits bank quad 4m (all 16
// lanes distinct) instead of the 16-way conflict a 4096B stride causes.
#define A_ROWS    32
#define LDS_LDA   (N_DIM + 8)                 // halves
#define LDS_BYTES (A_ROWS * LDS_LDA * 2)      // 131584 B

// -------------------------------------------------------------------------
// Pack (W_cc + C) into f16 B-fragments, where C is the circulant of the
// 5-tap kernel: C[k,n] = K_local[(k-n+2) mod N] when that index < 5.
// Folding the conv into the GEMM removes the entire 5-tap epilogue.
// Fragment = 32(K) x 16(N), contiguous-k B-layout (same k order as the A side): lanes 0-15 hold K=0..15 of column
// N=lane, lanes 16-31 hold K=16..31; 16 contiguous halves per lane.
// frag index = (k0/32)*128 + (n0/16); 8192 fragments, 32 lanes each.
// -------------------------------------------------------------------------
__global__ void pack_w_kernel(const float* __restrict__ W,
                              const float* __restrict__ Kl,
                              _Float16* __restrict__ Wp) {
    int tid  = blockIdx.x * blockDim.x + threadIdx.x;   // 262144 threads
    int lane = tid & 31;
    int frag = tid >> 5;                                // 0..8191
    int k0   = (frag >> 7) << 5;                        // (frag/128)*32
    int n0   = (frag & 127) << 4;                       // (frag%128)*16
    int kk   = k0 + ((lane >> 4) << 4);                 // +0 or +16
    int nn   = n0 + (lane & 15);
    _Float16* dst = Wp + ((size_t)frag << 9) + (lane << 4);
    const float* src = W + (size_t)kk * N_DIM + nn;
#pragma unroll
    for (int t = 0; t < 16; ++t) {
        const int d = ((kk + t) - nn + 2) & N_MASK;     // circulant band test
        const float add = (d < K_TAPS) ? Kl[d] : 0.0f;
        dst[t] = (_Float16)(src[(size_t)t * N_DIM] + add);
    }
}

// -------------------------------------------------------------------------
// Fused CML step:
//   stage:    logistic map of this block's 32 grid rows -> f16 A-tile in LDS
//   gemm:     coupled*2 = mapped @ (W_cc + C) via v_wmma_f32_16x16x32_f16,
//             A fragments via conflict-free ds_load_b128, B from L2
//   epilogue: eps/beta mixing + int8-style quantization (+clip on last step)
// Block: 128 threads = 4 waves, block tile 32(M) x 128(N), wave 32x32 (2x2).
// Grid (16, 8) covers 256x2048 exactly -> EXEC all-ones for every WMMA.
// -------------------------------------------------------------------------
__global__ __launch_bounds__(128)
void cml_step_kernel(const float*    __restrict__ gin,   // grid state [B][N] f32
                     const _Float16* __restrict__ Wp,    // packed B frags
                     const float*    __restrict__ rv,    // r[N]
                     const float*    __restrict__ eps,
                     const float*    __restrict__ beta,
                     const float*    __restrict__ drive,
                     float* __restrict__ gout,
                     int final_step) {
    extern __shared__ _Float16 Atile[];                  // [A_ROWS][LDS_LDA]

    const int tid    = threadIdx.x;
    const int lane   = tid & 31;
    const int wave   = tid >> 5;                         // 0..3 -> N position
    const int m0     = blockIdx.y * A_ROWS;
    const int n0     = blockIdx.x * 128 + wave * 32;
    const int laneM  = lane & 15;
    const int laneHi = lane >> 4;                        // 0 or 1

    // ---- stage: mapped = r * g * (1-g), f32 -> f16 into padded LDS -------
    for (int i = tid * 4; i < A_ROWS * N_DIM; i += 128 * 4) {
        const int row = i >> 11;                         // i / N_DIM
        const int col = i & N_MASK;
        const float4 g4 = *(const float4*)(gin + (size_t)(m0 + row) * N_DIM + col);
        const float4 r4 = *(const float4*)(rv + col);
        _Float16* d = Atile + row * LDS_LDA + col;       // 8B-aligned
        d[0] = (_Float16)(r4.x * g4.x * (1.0f - g4.x));
        d[1] = (_Float16)(r4.y * g4.y * (1.0f - g4.y));
        d[2] = (_Float16)(r4.z * g4.z * (1.0f - g4.z));
        d[3] = (_Float16)(r4.w * g4.w * (1.0f - g4.w));
    }
    __syncthreads();

    // ---- GEMM: A from LDS, B from global (L2-resident) -------------------
    // A 16x32 layout: lane m = laneM, 16 contiguous halves start at K + laneHi*16.
    const _Float16* a0base = Atile + laneM * LDS_LDA + (laneHi << 4);
    // Route the +16-row delta through an opaque offset so the compiler keeps
    // a second loop-invariant DS base register (all in-loop offsets are then
    // 16-bit immediates; no per-iteration VALU adds in WMMA hazard shadows).
    int a1off = 16 * LDS_LDA;                            // halves
    asm volatile("" : "+v"(a1off));
    const _Float16* a1base = a0base + a1off;
    // B fragment bases for the two N-tiles; +(k0/32)*128 frags per K step
    const _Float16* bbase0 = Wp + ((size_t)(n0 >> 4) << 9) + (lane << 4);
    const _Float16* bbase1 = bbase0 + 512;

    v8f c00 = {}, c01 = {}, c10 = {}, c11 = {};
    union Frag { v16h v; v8h h[2]; };

    for (int k0 = 0; k0 < N_DIM; k0 += 32) {
        const v8h* pa0 = (const v8h*)(a0base + k0);      // ds_load_b128 x2
        const v8h* pa1 = (const v8h*)(a1base + k0);
        const size_t bo = (size_t)(k0 >> 5) * (128 * 512);
        const v8h* pb0 = (const v8h*)(bbase0 + bo);      // global_load_b128 x2
        const v8h* pb1 = (const v8h*)(bbase1 + bo);

        Frag a0, a1, b0, b1;
        a0.h[0] = pa0[0]; a0.h[1] = pa0[1];              // K: +0..15
        a1.h[0] = pa1[0]; a1.h[1] = pa1[1];
        b0.h[0] = pb0[0]; b0.h[1] = pb0[1];
        b1.h[0] = pb1[0]; b1.h[1] = pb1[1];

        c00 = __builtin_amdgcn_wmma_f32_16x16x32_f16(false, a0.v, false, b0.v,
                                                     (short)0, c00, false, false);
        c01 = __builtin_amdgcn_wmma_f32_16x16x32_f16(false, a0.v, false, b1.v,
                                                     (short)0, c01, false, false);
        c10 = __builtin_amdgcn_wmma_f32_16x16x32_f16(false, a1.v, false, b0.v,
                                                     (short)0, c10, false, false);
        c11 = __builtin_amdgcn_wmma_f32_16x16x32_f16(false, a1.v, false, b1.v,
                                                     (short)0, c11, false, false);
    }

    // ---- epilogue (conv already inside the GEMM) -------------------------
    v8f cc[4] = {c00, c01, c10, c11};
#pragma unroll
    for (int t = 0; t < 4; ++t) {
        const int mtl = (t >> 1) << 4;                   // tile-local M base
        const int nn  = n0 + ((t & 1) << 4) + laneM;     // C layout: N = lane&15
        const float ep  = eps[nn];
        const float bt  = beta[nn];
        const float rn  = rv[nn];
        const float ome = 1.0f - ep;
        const float omb = 1.0f - bt;
#pragma unroll
        for (int v = 0; v < 8; ++v) {
            const int rowL = mtl + v + (laneHi << 3);    // C layout: M=v+8*(lane>=16)
            const size_t row = (size_t)(m0 + rowL) * N_DIM;
            const float gv      = gin[row + nn];         // direct term in f32
            const float mapped  = rn * gv * (1.0f - gv);
            const float coupled = 0.5f * cc[t][v];       // 0.5*(local+global)
            const float physics = ome * mapped + ep * coupled;
            float g = omb * physics + bt * drive[row + nn];
            g = rintf(g * 127.0f) / 127.0f;              // simulated int8
            if (final_step)
                g = fminf(fmaxf(g, 1.0e-4f), 1.0f - 1.0e-4f);
            gout[row + nn] = g;
        }
    }
}

// -------------------------------------------------------------------------
extern "C" void kernel_launch(void* const* d_in, const int* in_sizes, int n_in,
                              void* d_out, int out_size, void* d_ws, size_t ws_size,
                              hipStream_t stream) {
    (void)in_sizes; (void)n_in; (void)out_size; (void)ws_size;
    const float* drive = (const float*)d_in[0];
    const float* rv    = (const float*)d_in[1];
    const float* eps   = (const float*)d_in[2];
    const float* beta  = (const float*)d_in[3];
    const float* Kl    = (const float*)d_in[4];
    const float* Wcc   = (const float*)d_in[5];
    // d_in[6] = steps (64), d_in[7] = kernel_size (5): compile-time constants.

    char* ws = (char*)d_ws;
    _Float16* Wp = (_Float16*)(ws + OFF_WP);
    float*    gA = (float*)(ws + OFF_GA);
    float*    gB = (float*)(ws + OFF_GB);

    // one-time per launch: pack f16(W_cc + conv circulant) fragments
    pack_w_kernel<<<1024, 256, 0, stream>>>(Wcc, Kl, Wp);

    const float* cur = drive;                 // scan carry init = drive
    float* buf[2] = {gA, gB};
    const dim3 grid(N_DIM / 128, B_DIM / A_ROWS);   // (16, 8)
    for (int s = 0; s < N_STEPS; ++s) {
        float* nxt = (s == N_STEPS - 1) ? (float*)d_out : buf[s & 1];
        cml_step_kernel<<<grid, 128, LDS_BYTES, stream>>>(
            cur, Wp, rv, eps, beta, drive, nxt,
            (s == N_STEPS - 1) ? 1 : 0);
        cur = nxt;
    }
}
